// Net_44710609551887
// MI455X (gfx1250) — compile-verified, matrix operand order checked by reading, not run on hardware
//
#include <hip/hip_runtime.h>
#include <hip/hip_bf16.h>

// PointNet++ segmentation forward, fused for MI455X (gfx1250, wave32, WMMA).
// All GEMMs via v_wmma_f32_16x16x32_f16 with f32 accumulate; weights packed
// once to f16 in the per-lane WMMA B layout (L2-resident, 1x32B load/lane/tile).

typedef __attribute__((ext_vector_type(16))) _Float16 v16h;
typedef __attribute__((ext_vector_type(8)))  float    v8f;

#define B_  16
#define N_  8192
#define N1_ 328
#define N2_ 33
#define KNB 64
#define NEGV (-1e10f)

// ---------------- WMMA helpers ----------------
// A layout (16x32 f16): lane m=L&15; kbase = (L>=16)?8:0; halves 0..7 -> K=kbase+e,
// halves 8..15 -> K=kbase+16+e.  B packed identically per lane (pre-swizzled).
__device__ __forceinline__ v16h lds_load_a(const _Float16* A, int ld) {
    int lane = threadIdx.x & 31;
    const _Float16* row = A + (lane & 15) * ld + (((lane >> 4) & 1) << 3);
    v16h a;
#pragma unroll
    for (int e = 0; e < 8; ++e) a[e] = row[e];
#pragma unroll
    for (int e = 0; e < 8; ++e) a[8 + e] = row[16 + e];
    return a;
}

__device__ __forceinline__ v16h gld_b(const _Float16* Wp, int kt, int nt, int NT) {
    int lane = threadIdx.x & 31;
    const _Float16* p = Wp + (((size_t)(kt * NT + nt) * 32 + (size_t)lane) << 4);
    v16h b;
#pragma unroll
    for (int e = 0; e < 16; ++e) b[e] = p[e];
    return b;
}

__device__ __forceinline__ v8f gemm_tile(const _Float16* A, int Kp,
                                         const _Float16* Wp, int nt, int NT) {
    int KT = Kp >> 5;
    v8f acc = {};
    for (int kt = 0; kt < KT; ++kt) {
        v16h a = lds_load_a(A + kt * 32, Kp);
        v16h b = gld_b(Wp, kt, nt, NT);
        acc = __builtin_amdgcn_wmma_f32_16x16x32_f16(false, a, false, b,
                                                     (short)0, acc, false, false);
    }
    return acc;
}

// One 16-row layer: A[16 x Kp] (LDS,f16) -> Bout[16 x Coutp] (LDS,f16), ntiles [nt0,nt1)
__device__ __forceinline__ void gemm16(const _Float16* A, int Kp,
                                       const _Float16* Wp, const float* bias,
                                       int Cout, int Coutp, _Float16* Bout,
                                       int nt0, int nt1, bool relu) {
    int NT = Coutp >> 4;
    int lane = threadIdx.x & 31;
    int n = lane & 15, mb = ((lane >> 4) & 1) << 3;
    for (int nt = nt0; nt < nt1; ++nt) {
        v8f acc = gemm_tile(A, Kp, Wp, nt, NT);
        int col = nt * 16 + n;
        float bv = (col < Cout) ? bias[col] : 0.f;
#pragma unroll
        for (int v = 0; v < 8; ++v) {
            float x = acc[v] + bv;
            if (relu && x < 0.f) x = 0.f;
            Bout[(mb + v) * Coutp + col] = (_Float16)x;
        }
    }
}

// ---------------- Weight packing: W[Cin x Cout] f32 -> packed f16 B tiles ----------
__global__ void pack_w_kernel(const float* __restrict__ W, _Float16* __restrict__ out,
                              int Cin, int Cout) {
    int Kp = ((Cin + 31) / 32) * 32, Np2 = ((Cout + 15) / 16) * 16;
    int NT = Np2 / 16;
    int total = (Kp / 32) * NT * 32;
    for (int i = blockIdx.x * blockDim.x + threadIdx.x; i < total;
         i += gridDim.x * blockDim.x) {
        int lane = i & 31, tilei = i >> 5;
        int nt = tilei % NT, kt = tilei / NT;
        int n = nt * 16 + (lane & 15);
        int kb = kt * 32 + (((lane >> 4) & 1) << 3);
        _Float16* o = out + ((size_t)i << 4);
#pragma unroll
        for (int e = 0; e < 16; ++e) {
            int k = kb + ((e < 8) ? e : (8 + e));
            float v = (k < Cin && n < Cout) ? W[(size_t)k * Cout + n] : 0.f;
            o[e] = (_Float16)v;
        }
    }
}

// ---------------- FPS (one block per batch, LDS min-dist) ----------------
__global__ void fps_kernel(const float* __restrict__ pos, int Np, int S,
                           int* __restrict__ idx) {
    __shared__ float mds[N_];
    __shared__ float sv[256];
    __shared__ int   si[256];
    __shared__ float lp[3];
    int b = blockIdx.x, t = threadIdx.x;
    const float* P = pos + (size_t)b * Np * 3;
    int* out = idx + (size_t)b * S;
    for (int p = t; p < Np; p += 256) mds[p] = 1e10f;
    if (t == 0) { out[0] = 0; lp[0] = P[0]; lp[1] = P[1]; lp[2] = P[2]; }
    __syncthreads();
    for (int s = 1; s < S; ++s) {
        float lx = lp[0], ly = lp[1], lz = lp[2];
        float bv = -1.f; int bi = 0;
        for (int p = t; p < Np; p += 256) {
            float dx = P[p * 3] - lx, dy = P[p * 3 + 1] - ly, dz = P[p * 3 + 2] - lz;
            float m = fminf(mds[p], dx * dx + dy * dy + dz * dz);
            mds[p] = m;
            if (m > bv) { bv = m; bi = p; }
        }
        sv[t] = bv; si[t] = bi;
        __syncthreads();
        for (int o = 128; o > 0; o >>= 1) {
            if (t < o) {
                if (sv[t + o] > sv[t] || (sv[t + o] == sv[t] && si[t + o] < si[t])) {
                    sv[t] = sv[t + o]; si[t] = si[t + o];
                }
            }
            __syncthreads();
        }
        if (t == 0) {
            int w = si[0]; out[s] = w;
            lp[0] = P[w * 3]; lp[1] = P[w * 3 + 1]; lp[2] = P[w * 3 + 2];
        }
        __syncthreads();
    }
}

__global__ void gather3_kernel(const float* __restrict__ pos, const int* __restrict__ idx,
                               float* __restrict__ out, int Np, int S) {
    int i = blockIdx.x * blockDim.x + threadIdx.x;
    int total = B_ * S * 3;
    if (i >= total) return;
    int c = i % 3, rs = i / 3;
    int s = rs % S, b = rs / S;
    out[i] = pos[((size_t)b * Np + idx[b * S + s]) * 3 + c];
}

// ---------------- Ball query: K=64 nearest within radius (LDS d2 + 64 argmins) ------
__global__ void ball_kernel(const float* __restrict__ pos_src,
                            const float* __restrict__ centers,
                            int Np, int S, float r2, int* __restrict__ nbr) {
    __shared__ float d2s[N_];
    __shared__ float sv[256];
    __shared__ int   si[256];
    int id = blockIdx.x, b = id / S, s = id % S, t = threadIdx.x;
    const float* P = pos_src + (size_t)b * Np * 3;
    const float* C = centers + ((size_t)b * S + s) * 3;
    float cx = C[0], cy = C[1], cz = C[2];
    for (int p = t; p < Np; p += 256) {
        float dx = P[p * 3] - cx, dy = P[p * 3 + 1] - cy, dz = P[p * 3 + 2] - cz;
        float d = dx * dx + dy * dy + dz * dz;
        d2s[p] = (d <= r2) ? d : 3e38f;
    }
    __syncthreads();
    int* onbr = nbr + ((size_t)b * S + s) * KNB;
    for (int k = 0; k < KNB; ++k) {
        float bv = 3e38f; int bi = Np;
        for (int p = t; p < Np; p += 256) {
            float d = d2s[p];
            if (d < bv || (d == bv && p < bi)) { bv = d; bi = p; }
        }
        sv[t] = bv; si[t] = bi;
        __syncthreads();
        for (int o = 128; o > 0; o >>= 1) {
            if (t < o) {
                if (sv[t + o] < sv[t] || (sv[t + o] == sv[t] && si[t + o] < si[t])) {
                    sv[t] = sv[t + o]; si[t] = si[t + o];
                }
            }
            __syncthreads();
        }
        if (t == 0) {
            int w = (sv[0] < 2e38f) ? si[0] : -1;
            onbr[k] = w;
            if (w >= 0) d2s[w] = 3e38f;
        }
        __syncthreads();
    }
}

// ---------------- SA module: gather + 3-layer MLP (WMMA) + masked max-pool ----------
// block = 128 threads (4 waves); one center per block; wave w owns neighbors 16w..16w+15
__global__ void sa_mlp_kernel(const float* __restrict__ xsrc, int Cx,
                              const float* __restrict__ pos_src,
                              const float* __restrict__ centers,
                              const int* __restrict__ nbr, int Np, int S,
                              const _Float16* W1, const float* b1, int C1,
                              const _Float16* W2, const float* b2, int C2,
                              const _Float16* W3, const float* b3, int C3,
                              float* __restrict__ out) {
    __shared__ _Float16 Abuf[4][16 * 160];
    __shared__ _Float16 Bbuf[4][16 * 128];
    __shared__ float pmax[4][256];
    __shared__ int   vall[4][16];
    int id = blockIdx.x, b = id / S, s = id % S;
    int w = threadIdx.x >> 5, lane = threadIdx.x & 31;
    const float* C = centers + ((size_t)b * S + s) * 3;
    float cx = C[0], cy = C[1], cz = C[2];
    const int* nb = nbr + ((size_t)b * S + s) * KNB + w * 16;
    int Kp0 = ((Cx + 3 + 31) / 32) * 32;
    _Float16* A = Abuf[w];
    _Float16* Bf = Bbuf[w];
    if (lane < 16) vall[w][lane] = (nb[lane] >= 0);
    for (int i = lane; i < 16 * Kp0; i += 32) {
        int m = i / Kp0, c = i % Kp0;
        int n = nb[m];
        float v = 0.f;
        if (n >= 0) {
            if (c < Cx) v = xsrc[((size_t)b * Np + n) * Cx + c];
            else if (c < Cx + 3) {
                float cc = (c - Cx == 0) ? cx : ((c - Cx == 1) ? cy : cz);
                v = pos_src[((size_t)b * Np + n) * 3 + (c - Cx)] - cc;
            }
        }
        A[i] = (_Float16)v;
    }
    __syncthreads();
    gemm16(A, Kp0, W1, b1, C1, C1, Bf, 0, C1 / 16, true);
    __syncthreads();
    gemm16(Bf, C1, W2, b2, C2, C2, A, 0, C2 / 16, true);
    __syncthreads();
    int NT3 = C3 / 16;
    int n = lane & 15, mb = ((lane >> 4) & 1) << 3;
    for (int nt = 0; nt < NT3; ++nt) {
        v8f acc = gemm_tile(A, C2, W3, nt, NT3);
        int col = nt * 16 + n;
        float bv = b3[col];
        float vmax = NEGV;
#pragma unroll
        for (int v = 0; v < 8; ++v) {
            if (vall[w][mb + v]) {
                float x = acc[v] + bv;
                if (x < 0.f) x = 0.f;
                vmax = fmaxf(vmax, x);
            }
        }
        vmax = fmaxf(vmax, __shfl_xor(vmax, 16, 32));
        if (lane < 16) pmax[w][col] = vmax;
    }
    __syncthreads();
    for (int c = threadIdx.x; c < C3; c += 128) {
        float m0 = fmaxf(fmaxf(pmax[0][c], pmax[1][c]), fmaxf(pmax[2][c], pmax[3][c]));
        out[((size_t)b * S + s) * C3 + c] = m0;
    }
}

// ---------------- Global SA: MLP 259->256->512->1024 over 33 pts + global max -------
__global__ void sa3_kernel(const float* __restrict__ x2, const float* __restrict__ pos2,
                           const _Float16* W1, const float* b1,
                           const _Float16* W2, const float* b2,
                           const _Float16* W3, const float* b3,
                           float* __restrict__ xg) {
    __shared__ _Float16 A[16 * 512];
    __shared__ _Float16 Bf[16 * 512];
    __shared__ float gmax[1024];
    int b = blockIdx.x, t = threadIdx.x, w = t >> 5, lane = t & 31;
    for (int c = t; c < 1024; c += 128) gmax[c] = NEGV;
    __syncthreads();
    for (int tile = 0; tile < 3; ++tile) {
        for (int i = t; i < 16 * 288; i += 128) {
            int m = i / 288, c = i % 288, r = tile * 16 + m;
            float v = 0.f;
            if (r < N2_) {
                if (c < 256) v = x2[((size_t)b * N2_ + r) * 256 + c];
                else if (c < 259) v = pos2[((size_t)b * N2_ + r) * 3 + (c - 256)];
            }
            A[m * 288 + c] = (_Float16)v;
        }
        __syncthreads();
        gemm16(A, 288, W1, b1, 256, 256, Bf, w * 4, w * 4 + 4, true);
        __syncthreads();
        gemm16(Bf, 256, W2, b2, 512, 512, A, w * 8, w * 8 + 8, true);
        __syncthreads();
        int n = lane & 15, mb = ((lane >> 4) & 1) << 3;
        for (int nt = w * 16; nt < w * 16 + 16; ++nt) {
            v8f acc = gemm_tile(A, 512, W3, nt, 64);
            int col = nt * 16 + n;
            float bv = b3[col];
            float vmax = NEGV;
#pragma unroll
            for (int v = 0; v < 8; ++v) {
                int r = tile * 16 + mb + v;
                if (r < N2_) {
                    float x = acc[v] + bv;
                    if (x < 0.f) x = 0.f;
                    vmax = fmaxf(vmax, x);
                }
            }
            vmax = fmaxf(vmax, __shfl_xor(vmax, 16, 32));
            if (lane < 16) gmax[col] = fmaxf(gmax[col], vmax);
        }
        __syncthreads();
    }
    for (int c = t; c < 1024; c += 128) xg[(size_t)b * 1024 + c] = gmax[c];
}

// ---------------- FP3: concat(xg bcast, x2) -> 1280->256->256 ----------------------
__global__ void fp3_kernel(const float* __restrict__ xg, const float* __restrict__ x2,
                           const _Float16* W1, const float* b1,
                           const _Float16* W2, const float* b2,
                           float* __restrict__ x2u) {
    __shared__ _Float16 A[16 * 1280];
    __shared__ _Float16 Bf[16 * 256];
    int b = blockIdx.x, t = threadIdx.x, w = t >> 5;
    for (int tile = 0; tile < 3; ++tile) {
        for (int i = t; i < 16 * 1280; i += 128) {
            int m = i / 1280, c = i % 1280, r = tile * 16 + m;
            float v = 0.f;
            if (r < N2_)
                v = (c < 1024) ? xg[(size_t)b * 1024 + c]
                               : x2[((size_t)b * N2_ + r) * 256 + (c - 1024)];
            A[i] = (_Float16)v;
        }
        __syncthreads();
        gemm16(A, 1280, W1, b1, 256, 256, Bf, w * 4, w * 4 + 4, true);
        __syncthreads();
        gemm16(Bf, 256, W2, b2, 256, 256, A, w * 4, w * 4 + 4, true);
        __syncthreads();
        for (int i = t; i < 16 * 256; i += 128) {
            int m = i / 256, c = i % 256, r = tile * 16 + m;
            if (r < N2_) x2u[((size_t)b * N2_ + r) * 256 + c] = (float)A[m * 256 + c];
        }
        __syncthreads();
    }
}

// ---------------- kNN(3) helper: top-3 nearest among Ns sources --------------------
__device__ __forceinline__ void knn3(const float* __restrict__ src_pos, int Ns,
                                     float tx, float ty, float tz,
                                     int* bi, float* bw, float* wsum) {
    float bd0 = 3e38f, bd1 = 3e38f, bd2 = 3e38f;
    int i0 = 0, i1 = 0, i2 = 0;
    for (int s = 0; s < Ns; ++s) {
        float dx = src_pos[s * 3] - tx, dy = src_pos[s * 3 + 1] - ty,
              dz = src_pos[s * 3 + 2] - tz;
        float d = dx * dx + dy * dy + dz * dz;
        if (d < bd0) { bd2 = bd1; i2 = i1; bd1 = bd0; i1 = i0; bd0 = d; i0 = s; }
        else if (d < bd1) { bd2 = bd1; i2 = i1; bd1 = d; i1 = s; }
        else if (d < bd2) { bd2 = d; i2 = s; }
    }
    float w0 = 1.f / fmaxf(bd0, 1e-16f);
    float w1 = 1.f / fmaxf(bd1, 1e-16f);
    float w2 = 1.f / fmaxf(bd2, 1e-16f);
    bi[0] = i0; bi[1] = i1; bi[2] = i2;
    bw[0] = w0; bw[1] = w1; bw[2] = w2;
    *wsum = w0 + w1 + w2;
}

// ---------------- FP2: interp(33->328,k=3) + MLP 384->256->128 (1 wave/tile) -------
__global__ void fp2_kernel(const float* __restrict__ x2u, const float* __restrict__ pos2,
                           const float* __restrict__ pos1, const float* __restrict__ x1,
                           const _Float16* W1, const float* b1,
                           const _Float16* W2, const float* b2,
                           float* __restrict__ x1u) {
    __shared__ _Float16 A[16 * 384];
    __shared__ _Float16 Bf[16 * 256];
    __shared__ int   sidx[16][3];
    __shared__ float swt[16][3];
    __shared__ float swsum[16];
    int blk = blockIdx.x, b = blk / 21, tile = blk % 21;
    int lane = threadIdx.x;
    if (lane < 16) {
        int r = tile * 16 + lane;
        if (r < N1_) {
            const float* tp = pos1 + ((size_t)b * N1_ + r) * 3;
            float ws;
            knn3(pos2 + (size_t)b * N2_ * 3, N2_, tp[0], tp[1], tp[2],
                 sidx[lane], swt[lane], &ws);
            swsum[lane] = ws;
        } else {
            sidx[lane][0] = sidx[lane][1] = sidx[lane][2] = 0;
            swt[lane][0] = swt[lane][1] = swt[lane][2] = 0.f;
            swsum[lane] = 1.f;
        }
    }
    __syncthreads();
    for (int i = lane; i < 16 * 384; i += 32) {
        int m = i / 384, c = i % 384, r = tile * 16 + m;
        float v = 0.f;
        if (r < N1_) {
            if (c < 256) {
                float acc = 0.f;
#pragma unroll
                for (int j = 0; j < 3; ++j)
                    acc += swt[m][j] * x2u[((size_t)b * N2_ + sidx[m][j]) * 256 + c];
                v = acc / swsum[m];
            } else v = x1[((size_t)b * N1_ + r) * 128 + (c - 256)];
        }
        A[i] = (_Float16)v;
    }
    __syncthreads();
    gemm16(A, 384, W1, b1, 256, 256, Bf, 0, 16, true);
    __syncthreads();
    int n = lane & 15, mb = ((lane >> 4) & 1) << 3;
    for (int nt = 0; nt < 8; ++nt) {
        v8f acc = gemm_tile(Bf, 256, W2, nt, 8);
        int col = nt * 16 + n;
        float bv = b2[col];
#pragma unroll
        for (int v = 0; v < 8; ++v) {
            int r = tile * 16 + mb + v;
            if (r < N1_) {
                float x = acc[v] + bv;
                if (x < 0.f) x = 0.f;
                x1u[((size_t)b * N1_ + r) * 128 + col] = x;
            }
        }
    }
}

// ------- FP1 + head fused: interp(328->8192,k=3) + 6 GEMM layers + log_softmax -----
__global__ void fp1_head_kernel(const float* __restrict__ x, const float* __restrict__ pos,
                                const float* __restrict__ pos1, const float* __restrict__ x1u,
                                const _Float16* Wf1, const float* bf1,
                                const _Float16* Wf2, const float* bf2,
                                const _Float16* Wf3, const float* bf3,
                                const _Float16* Wl1, const float* bl1,
                                const _Float16* Wl2, const float* bl2,
                                const _Float16* Wl3, const float* bl3,
                                float* __restrict__ out) {
    __shared__ _Float16 A0[16 * 160];
    __shared__ _Float16 A1[16 * 128];
    __shared__ float H[16 * 64];
    __shared__ int   sidx[16][3];
    __shared__ float swt[16][3];
    __shared__ float swsum[16];
    int blk = blockIdx.x, b = blk >> 9, tile = blk & 511;
    int lane = threadIdx.x;
    if (lane < 16) {
        int r = tile * 16 + lane;
        const float* tp = pos + ((size_t)b * N_ + r) * 3;
        float ws;
        knn3(pos1 + (size_t)b * N1_ * 3, N1_, tp[0], tp[1], tp[2],
             sidx[lane], swt[lane], &ws);
        swsum[lane] = ws;
    }
    __syncthreads();
    for (int i = lane; i < 16 * 160; i += 32) {
        int m = i / 160, c = i % 160;
        float v = 0.f;
        if (c < 128) {
            float acc = 0.f;
#pragma unroll
            for (int j = 0; j < 3; ++j)
                acc += swt[m][j] * x1u[((size_t)b * N1_ + sidx[m][j]) * 128 + c];
            v = acc / swsum[m];
        } else if (c < 131) {
            v = x[((size_t)b * N_ + tile * 16 + m) * 3 + (c - 128)];
        }
        A0[i] = (_Float16)v;
    }
    __syncthreads();
    gemm16(A0, 160, Wf1, bf1, 128, 128, A1, 0, 8, true);  __syncthreads();
    gemm16(A1, 128, Wf2, bf2, 128, 128, A0, 0, 8, true);  __syncthreads();
    gemm16(A0, 128, Wf3, bf3, 128, 128, A1, 0, 8, true);  __syncthreads();
    gemm16(A1, 128, Wl1, bl1, 128, 128, A0, 0, 8, true);  __syncthreads();
    gemm16(A0, 128, Wl2, bl2, 128, 128, A1, 0, 8, false); __syncthreads();
    int n = lane & 15, mb = ((lane >> 4) & 1) << 3;
    for (int nt = 0; nt < 4; ++nt) {
        v8f acc = gemm_tile(A1, 128, Wl3, nt, 4);
        int col = nt * 16 + n;
        float bv = (col < 50) ? bl3[col] : 0.f;
#pragma unroll
        for (int v = 0; v < 8; ++v) H[(mb + v) * 64 + col] = acc[v] + bv;
    }
    __syncthreads();
    if (lane < 16) {
        int r = lane;
        float mx = -3e38f;
        for (int c = 0; c < 50; ++c) mx = fmaxf(mx, H[r * 64 + c]);
        float sum = 0.f;
        for (int c = 0; c < 50; ++c) sum += expf(H[r * 64 + c] - mx);
        float lse = mx + logf(sum);
        size_t base = ((size_t)b * N_ + tile * 16 + r) * 50;
        for (int c = 0; c < 50; ++c) out[base + c] = H[r * 64 + c] - lse;
    }
}

// ==================================================================================
extern "C" void kernel_launch(void* const* d_in, const int* in_sizes, int n_in,
                              void* d_out, int out_size, void* d_ws, size_t ws_size,
                              hipStream_t stream) {
    (void)in_sizes; (void)n_in; (void)out_size; (void)ws_size;
    char* WS = (char*)d_ws;
    size_t off = 0;
    auto alloc = [&](size_t bytes) -> size_t {
        size_t o = off; off += (bytes + 255) & ~(size_t)255; return o;
    };
    auto pbytes = [](int cin, int cout) -> size_t {
        int Kp = ((cin + 31) / 32) * 32, Np2 = ((cout + 15) / 16) * 16;
        return (size_t)Kp * Np2 * 2;
    };

    // ---- packed weight offsets (19 matrices, order below) ----
    const int widx[19]  = {2, 4, 6, 8, 10, 12, 14, 16, 18, 20, 22, 24, 26, 28, 30, 32, 34, 36, 38};
    const int wcin[19]  = {6, 64, 64, 131, 128, 128, 259, 256, 512, 1280, 256, 384, 256, 131, 128, 128, 128, 128, 128};
    const int wcout[19] = {64, 64, 128, 128, 128, 256, 256, 512, 1024, 256, 256, 256, 128, 128, 128, 128, 128, 128, 50};
    size_t woff[19];
    for (int i = 0; i < 19; ++i) woff[i] = alloc(pbytes(wcin[i], wcout[i]));

    // ---- intermediate buffers ----
    size_t o_idx1 = alloc((size_t)B_ * N1_ * 4);
    size_t o_pos1 = alloc((size_t)B_ * N1_ * 3 * 4);
    size_t o_nbr1 = alloc((size_t)B_ * N1_ * KNB * 4);
    size_t o_x1   = alloc((size_t)B_ * N1_ * 128 * 4);
    size_t o_idx2 = alloc((size_t)B_ * N2_ * 4);
    size_t o_pos2 = alloc((size_t)B_ * N2_ * 3 * 4);
    size_t o_nbr2 = alloc((size_t)B_ * N2_ * KNB * 4);
    size_t o_x2   = alloc((size_t)B_ * N2_ * 256 * 4);
    size_t o_xg   = alloc((size_t)B_ * 1024 * 4);
    size_t o_x2u  = alloc((size_t)B_ * N2_ * 256 * 4);
    size_t o_x1u  = alloc((size_t)B_ * N1_ * 128 * 4);

    // ---- pack all weights to f16 WMMA layout ----
    for (int i = 0; i < 19; ++i) {
        int Kp = ((wcin[i] + 31) / 32) * 32, Np2 = ((wcout[i] + 15) / 16) * 16;
        int total = (Kp / 32) * (Np2 / 16) * 32;
        int grid = (total + 255) / 256;
        pack_w_kernel<<<grid, 256, 0, stream>>>((const float*)d_in[widx[i]],
                                                (_Float16*)(WS + woff[i]),
                                                wcin[i], wcout[i]);
    }
    auto PW = [&](int i) { return (const _Float16*)(WS + woff[i]); };
    auto BI = [&](int i) { return (const float*)d_in[i]; };

    const float* x   = (const float*)d_in[0];
    const float* pos = (const float*)d_in[1];
    int*   idx1 = (int*)(WS + o_idx1);
    float* pos1 = (float*)(WS + o_pos1);
    int*   nbr1 = (int*)(WS + o_nbr1);
    float* x1   = (float*)(WS + o_x1);
    int*   idx2 = (int*)(WS + o_idx2);
    float* pos2 = (float*)(WS + o_pos2);
    int*   nbr2 = (int*)(WS + o_nbr2);
    float* x2   = (float*)(WS + o_x2);
    float* xg   = (float*)(WS + o_xg);
    float* x2u  = (float*)(WS + o_x2u);
    float* x1u  = (float*)(WS + o_x1u);
    float* out  = (float*)d_out;

    // ---- SA1 ----
    fps_kernel<<<B_, 256, 0, stream>>>(pos, N_, N1_, idx1);
    gather3_kernel<<<(B_ * N1_ * 3 + 255) / 256, 256, 0, stream>>>(pos, idx1, pos1, N_, N1_);
    ball_kernel<<<B_ * N1_, 256, 0, stream>>>(pos, pos1, N_, N1_, 1.0f, nbr1);
    sa_mlp_kernel<<<B_ * N1_, 128, 0, stream>>>(x, 3, pos, pos1, nbr1, N_, N1_,
        PW(0), BI(3), 64, PW(1), BI(5), 64, PW(2), BI(7), 128, x1);
    // ---- SA2 ----
    fps_kernel<<<B_, 256, 0, stream>>>(pos1, N1_, N2_, idx2);
    gather3_kernel<<<(B_ * N2_ * 3 + 255) / 256, 256, 0, stream>>>(pos1, idx2, pos2, N1_, N2_);
    ball_kernel<<<B_ * N2_, 256, 0, stream>>>(pos1, pos2, N1_, N2_, 25.0f, nbr2);
    sa_mlp_kernel<<<B_ * N2_, 128, 0, stream>>>(x1, 128, pos1, pos2, nbr2, N1_, N2_,
        PW(3), BI(9), 128, PW(4), BI(11), 128, PW(5), BI(13), 256, x2);
    // ---- Global SA ----
    sa3_kernel<<<B_, 128, 0, stream>>>(x2, pos2, PW(6), BI(15), PW(7), BI(17),
                                       PW(8), BI(19), xg);
    // ---- FP3 / FP2 / FP1+head ----
    fp3_kernel<<<B_, 128, 0, stream>>>(xg, x2, PW(9), BI(21), PW(10), BI(23), x2u);
    fp2_kernel<<<B_ * 21, 32, 0, stream>>>(x2u, pos2, pos1, x1,
                                           PW(11), BI(25), PW(12), BI(27), x1u);
    fp1_head_kernel<<<B_ * (N_ / 16), 32, 0, stream>>>(x, pos, pos1, x1u,
        PW(13), BI(29), PW(14), BI(31), PW(15), BI(33),
        PW(16), BI(35), PW(17), BI(37), PW(18), BI(39), out);
}
